// VEncoder_18056042512862
// MI455X (gfx1250) — matrix operand order repaired and checked by reading, not compiled, not run on hardware
//
#include <hip/hip_runtime.h>

#define NF 512   // input features
#define NH 256   // hidden
#define LAT 64   // latent

typedef __attribute__((ext_vector_type(16))) __bf16 v16bf;
typedef __attribute__((ext_vector_type(8)))  float  v8f;
typedef unsigned int u32x4 __attribute__((ext_vector_type(4)));
typedef int          i32x8 __attribute__((ext_vector_type(8)));
typedef int          i32x4 __attribute__((ext_vector_type(4)));

__device__ __forceinline__ unsigned short f2bf(float f) {
    unsigned u = __builtin_bit_cast(unsigned, f);
    u += 0x7FFFu + ((u >> 16) & 1u);           // round-to-nearest-even
    return (unsigned short)(u >> 16);
}

// ---------------------------------------------------------------------------
// Tensor Data Mover: async 2D tile load (bf16 elements) global -> LDS.
// Deposits tile rows with 16B padding after each 64B row => LDS pitch 80B
// (40 elements), matching the WMMA fragment addressing below.
// D# layout per CDNA5 ISA. rem0/rem1 are the remaining
// tensor extent from the tile origin (OOB reads return zero => free M guard).
// This toolchain's builtin takes 6 args (g0, g1, g2, g3, g4, cpol).
// ---------------------------------------------------------------------------
__device__ __forceinline__ void tdm_load_2d(
    const unsigned short* gtile,   // global address of tile start
    unsigned lds_off,              // LDS byte offset of destination
    unsigned rem0,                 // remaining elements along dim0 (K)
    unsigned rem1,                 // remaining rows along dim1
    unsigned tile0,                // tile width  (elements)
    unsigned tile1,                // tile height (rows)
    unsigned stride0)              // tensor row stride (elements)
{
    unsigned long long ga = (unsigned long long)(size_t)gtile;
    u32x4 g0;
    g0[0] = 1u;                                    // count=1, user descriptor
    g0[1] = lds_off;                               // lds_addr (bytes)
    g0[2] = (unsigned)ga;                          // global_addr[31:0]
    g0[3] = (unsigned)(ga >> 32) | 0x80000000u;    // global_addr[56:32] | type=2
    i32x8 g1;
    g1[0] = (int)((1u << 16)                       // data_size = 2 bytes
                | (1u << 20)                       // pad_enable
                | (3u << 22)                       // pad_interval: 16 DW (64B)
                | (3u << 25));                     // pad_amount:   4 DW (16B)
    g1[1] = (int)((rem0 & 0xFFFFu) << 16);                               // tensor_dim0 lo
    g1[2] = (int)(((rem0 >> 16) & 0xFFFFu) | ((rem1 & 0xFFFFu) << 16));  // dim0 hi | dim1 lo
    g1[3] = (int)(((rem1 >> 16) & 0xFFFFu) | (tile0 << 16));             // dim1 hi | tile_dim0
    g1[4] = (int)tile1;                            // tile_dim1 (tile_dim2 = 0)
    g1[5] = (int)stride0;                          // tensor_dim0_stride[31:0]
    g1[6] = 0;
    g1[7] = 0;
    i32x4 gz4 = {0, 0, 0, 0};
    i32x8 gz8 = {0, 0, 0, 0, 0, 0, 0, 0};
    __builtin_amdgcn_tensor_load_to_lds(g0, g1, gz4, gz4, gz8, 0);
}

// ---------------- utility kernels ----------------

__global__ void zero_f32(float* __restrict__ p, long long n) {
    long long i = (long long)blockIdx.x * blockDim.x + threadIdx.x;
    long long s = (long long)gridDim.x * blockDim.x;
    for (; i < n; i += s) p[i] = 0.0f;
}

__global__ void convert_bf16_vec4(const float* __restrict__ in,
                                  unsigned short* __restrict__ out, long long n4) {
    long long i = (long long)blockIdx.x * blockDim.x + threadIdx.x;
    long long s = (long long)gridDim.x * blockDim.x;
    for (; i < n4; i += s) {
        float4 v = ((const float4*)in)[i];
        ushort4 r;
        r.x = f2bf(v.x); r.y = f2bf(v.y); r.z = f2bf(v.z); r.w = f2bf(v.w);
        ((ushort4*)out)[i] = r;
    }
}

// W1 [NF][NH] f32 -> W1T [NH][NF] bf16 (transposed)
__global__ void conv_w1_t(const float* __restrict__ W1, unsigned short* __restrict__ W1T) {
    int i = blockIdx.x * blockDim.x + threadIdx.x;          // i over NH*NF
    if (i >= NH * NF) return;
    int n = i >> 9;          // / NF
    int k = i & (NF - 1);
    W1T[i] = f2bf(W1[(size_t)k * NH + n]);
}

// [W_mu | W_logvar] -> WCT [2*LAT][NH] bf16 (transposed)
__global__ void conv_wcat_t(const float* __restrict__ Wmu, const float* __restrict__ Wlv,
                            unsigned short* __restrict__ WCT) {
    int i = blockIdx.x * blockDim.x + threadIdx.x;          // i over (2*LAT)*NH
    if (i >= 2 * LAT * NH) return;
    int n = i >> 8;          // / NH
    int k = i & (NH - 1);
    float v = (n < LAT) ? Wmu[(size_t)k * LAT + n] : Wlv[(size_t)k * LAT + (n - LAT)];
    WCT[i] = f2bf(v);
}

// ---------------------------------------------------------------------------
// WMMA GEMM: C[M][N] = A[M][K](bf16) * Bt[N][K](bf16)^T
// Block tile 128(M) x 64(N), 256 threads = 8 waves stacked along M.
// Each wave: one A fragment reused across 4 WMMA tiles (full 64-wide N).
// Global -> LDS staging via TDM (tensor_load_to_lds), double buffered on
// TENSORcnt, pad-deposited at PITCH=40 elements.
// ---------------------------------------------------------------------------
__global__ __launch_bounds__(256) void gemm_bf16_wmma(
    const unsigned short* __restrict__ A,    // [M][K] bf16 row-major
    const unsigned short* __restrict__ Bt,   // [N][K] bf16 row-major (pre-transposed)
    float* __restrict__ C,                   // [M][N] f32
    int M, int N, int K)
{
    constexpr int BM = 128, BN = 64, PITCH = 40;   // 80B LDS pitch (16B aligned rows)
    __shared__ alignas(16) unsigned short sA[2][BM * PITCH];
    __shared__ alignas(16) unsigned short sB[2][BN * PITCH];

    const int tid  = threadIdx.x;
    const int lane = tid & 31;
    const int wave = tid >> 5;               // 0..7 : M sub-tile
    const int half = lane >> 4;              // 0/1
    const int l16  = lane & 15;

    const int m_blk = blockIdx.x * BM;
    const int n_blk = blockIdx.y * BN;

    const unsigned ldsA[2] = { (unsigned)(size_t)&sA[0][0], (unsigned)(size_t)&sA[1][0] };
    const unsigned ldsB[2] = { (unsigned)(size_t)&sB[0][0], (unsigned)(size_t)&sB[1][0] };

    v8f acc[4] = {};

    // prime buffer 0
    if (wave == 0) {
        tdm_load_2d(A  + (size_t)m_blk * K, ldsA[0], (unsigned)K, (unsigned)(M - m_blk),
                    32u, (unsigned)BM, (unsigned)K);
        tdm_load_2d(Bt + (size_t)n_blk * K, ldsB[0], (unsigned)K, (unsigned)(N - n_blk),
                    32u, (unsigned)BN, (unsigned)K);
    }

    int cur = 0;
    for (int k0 = 0; k0 < K; k0 += 32) {
        if (wave == 0) {
            if (k0 + 32 < K) {
                int kn = k0 + 32;
                tdm_load_2d(A  + (size_t)m_blk * K + kn, ldsA[cur ^ 1],
                            (unsigned)(K - kn), (unsigned)(M - m_blk), 32u, (unsigned)BM,
                            (unsigned)K);
                tdm_load_2d(Bt + (size_t)n_blk * K + kn, ldsB[cur ^ 1],
                            (unsigned)(K - kn), (unsigned)(N - n_blk), 32u, (unsigned)BN,
                            (unsigned)K);
                __builtin_amdgcn_s_wait_tensorcnt(2);   // current pair complete
            } else {
                __builtin_amdgcn_s_wait_tensorcnt(0);
            }
        }
        __syncthreads();                                // tiles visible to all waves

        // A fragment, 16x32 bf16 per CDNA5 layout:
        //  lanes 0-15: K0-7 (v0-3), K16-23 (v4-7); lanes 16-31: K8-15, K24-31
        union Frag { v16bf v; uint4 q[2]; } af, bfr;
        const unsigned short* ap = sA[cur] + (wave * 16 + l16) * PITCH;
        af.q[0] = *(const uint4*)(ap + half * 8);
        af.q[1] = *(const uint4*)(ap + 16 + half * 8);

#pragma unroll
        for (int t = 0; t < 4; ++t) {
            // B fragment, 32x16: lanes 0-15 hold K0-7, K16-23 of col n, lanes 16-31 K8-15, K24-31
            const unsigned short* bp = sB[cur] + (t * 16 + l16) * PITCH + half * 8;
            bfr.q[0] = *(const uint4*)(bp);
            bfr.q[1] = *(const uint4*)(bp + 16);
            acc[t] = __builtin_amdgcn_wmma_f32_16x16x32_bf16(false, af.v, false, bfr.v,
                                                             (short)0, acc[t], false, false);
        }
        __syncthreads();                                // protect buffer before next overwrite
        cur ^= 1;
    }

    // C layout: VGPR r -> M = r + 8*(lane>=16), N = lane&15
    const int m_base = m_blk + wave * 16 + half * 8;
#pragma unroll
    for (int t = 0; t < 4; ++t) {
        const int n0 = n_blk + t * 16 + l16;
#pragma unroll
        for (int r = 0; r < 8; ++r) {
            int mg = m_base + r;
            if (mg < M) C[(size_t)mg * N + n0] = acc[t][r];
        }
    }
}

// ---------------- SpMM scatter: out[dst] += w_e * feat[src] ----------------

__global__ __launch_bounds__(256) void spmm_scatter(
    const int* __restrict__ src, const int* __restrict__ dst,
    const float* __restrict__ w, const float* __restrict__ feat,
    float* __restrict__ out, long long total, int lg /* log2(D/4) */, int D)
{
    long long i = (long long)blockIdx.x * blockDim.x + threadIdx.x;
    long long s = (long long)gridDim.x * blockDim.x;
    for (; i < total; i += s) {
        long long e = i >> lg;
        int c = (int)(i & ((1LL << lg) - 1)) * 4;
        int sn = src[e], dn = dst[e];
        float we = w[e];
        float4 v = *(const float4*)(feat + (size_t)sn * D + c);
        float* o = out + (size_t)dn * D + c;
        atomicAdd(o + 0, we * v.x);
        atomicAdd(o + 1, we * v.y);
        atomicAdd(o + 2, we * v.z);
        atomicAdd(o + 3, we * v.w);
    }
}

// SpMM scatter from z[M][128] into split output: cols 0-63 -> mu, 64-127 -> logvar
__global__ __launch_bounds__(256) void spmm_scatter_split(
    const int* __restrict__ src, const int* __restrict__ dst,
    const float* __restrict__ w, const float* __restrict__ z,
    float* __restrict__ out, long long total, long long M)
{
    long long i = (long long)blockIdx.x * blockDim.x + threadIdx.x;
    long long s = (long long)gridDim.x * blockDim.x;
    for (; i < total; i += s) {
        long long e = i >> 5;                // 32 float4 chunks per edge (D=128)
        int c = (int)(i & 31) * 4;
        int sn = src[e], dn = dst[e];
        float we = w[e];
        float4 v = *(const float4*)(z + (size_t)sn * (2 * LAT) + c);
        float* o = (c < LAT) ? out + (size_t)dn * LAT + c
                             : out + (size_t)M * LAT + (size_t)dn * LAT + (c - LAT);
        atomicAdd(o + 0, we * v.x);
        atomicAdd(o + 1, we * v.y);
        atomicAdd(o + 2, we * v.z);
        atomicAdd(o + 3, we * v.w);
    }
}

// hb = bf16(relu(h + b1)), vectorized by 4
__global__ void bias_relu_bf16(const float* __restrict__ h, const float* __restrict__ b,
                               unsigned short* __restrict__ hb, long long n4) {
    long long i = (long long)blockIdx.x * blockDim.x + threadIdx.x;
    long long s = (long long)gridDim.x * blockDim.x;
    for (; i < n4; i += s) {
        int j = (int)((i * 4) & (NH - 1));
        float4 v = ((const float4*)h)[i];
        v.x += b[j + 0]; v.y += b[j + 1]; v.z += b[j + 2]; v.w += b[j + 3];
        ushort4 r;
        r.x = f2bf(v.x > 0.f ? v.x : 0.f);
        r.y = f2bf(v.y > 0.f ? v.y : 0.f);
        r.z = f2bf(v.z > 0.f ? v.z : 0.f);
        r.w = f2bf(v.w > 0.f ? v.w : 0.f);
        ((ushort4*)hb)[i] = r;
    }
}

// out[:half] += b_mu, out[half:] += b_logvar
__global__ void bias_out(float* __restrict__ out, const float* __restrict__ bmu,
                         const float* __restrict__ blv, long long half) {
    long long i = (long long)blockIdx.x * blockDim.x + threadIdx.x;
    long long s = (long long)gridDim.x * blockDim.x;
    for (; i < 2 * half; i += s) {
        int f = (int)(i & (LAT - 1));
        out[i] += (i < half) ? bmu[f] : blv[f];
    }
}

// ---------------- launch ----------------

extern "C" void kernel_launch(void* const* d_in, const int* in_sizes, int n_in,
                              void* d_out, int out_size, void* d_ws, size_t ws_size,
                              hipStream_t stream) {
    const float* x   = (const float*)d_in[0];
    const int*   ei  = (const int*)d_in[1];
    const float* ew  = (const float*)d_in[2];
    const float* W1  = (const float*)d_in[3];
    const float* b1  = (const float*)d_in[4];
    const float* Wmu = (const float*)d_in[5];
    const float* bmu = (const float*)d_in[6];
    const float* Wlv = (const float*)d_in[7];
    const float* blv = (const float*)d_in[8];
    float* out = (float*)d_out;

    const long long M = (long long)in_sizes[0] / NF;   // 100000
    const long long E = (long long)in_sizes[1] / 2;    // 3200000
    const int* src = ei;
    const int* dst = ei + E;

    // workspace layout (H aliases XB: XB dead after GEMM1, both 102.4 MB)
    char* ws = (char*)d_ws;
    size_t off = 0;
    auto take = [&](size_t bytes) { size_t o = off; off = (off + bytes + 255) & ~(size_t)255; return o; };
    float*          XW  = (float*)(ws + take((size_t)M * NH * 4));
    unsigned short* XB  = (unsigned short*)(ws + take((size_t)M * NF * 2));
    float*          H   = (float*)XB;
    unsigned short* HB  = (unsigned short*)(ws + take((size_t)M * NH * 2));
    float*          Z   = (float*)(ws + take((size_t)M * 2 * LAT * 4));
    unsigned short* W1T = (unsigned short*)(ws + take((size_t)NH * NF * 2));
    unsigned short* WCT = (unsigned short*)(ws + take((size_t)2 * LAT * NH * 2));
    (void)n_in; (void)out_size; (void)ws_size;

    // 1. weight conversions (transposed bf16)
    conv_w1_t<<<(NH * NF + 255) / 256, 256, 0, stream>>>(W1, W1T);
    conv_wcat_t<<<(2 * LAT * NH + 255) / 256, 256, 0, stream>>>(Wmu, Wlv, WCT);

    // 2. x -> bf16
    {
        long long n4 = M * NF / 4;
        convert_bf16_vec4<<<(int)((n4 + 255) / 256), 256, 0, stream>>>(x, XB, n4);
    }

    // 3. GEMM1: XW = x @ W1
    {
        dim3 grid((unsigned)((M + 127) / 128), NH / 64);
        gemm_bf16_wmma<<<grid, 256, 0, stream>>>(XB, W1T, XW, (int)M, NH, NF);
    }

    // 4. zero H (aliased over XB), then SpMM1 scatter
    {
        long long n = M * NH;
        zero_f32<<<(int)((n + 255) / 256), 256, 0, stream>>>(H, n);
        long long total = E * (NH / 4);
        spmm_scatter<<<(int)((total + 255) / 256), 256, 0, stream>>>(
            src, dst, ew, XW, H, total, 6 /* log2(64) */, NH);
    }

    // 5. bias + relu -> bf16
    {
        long long n4 = M * NH / 4;
        bias_relu_bf16<<<(int)((n4 + 255) / 256), 256, 0, stream>>>(H, b1, HB, n4);
    }

    // 6. GEMM2: Z = h @ [W_mu | W_logvar]
    {
        dim3 grid((unsigned)((M + 127) / 128), (2 * LAT) / 64);
        gemm_bf16_wmma<<<grid, 256, 0, stream>>>(HB, WCT, Z, (int)M, 2 * LAT, NH);
    }

    // 7. zero output, SpMM2 scatter split, add biases
    {
        long long n = 2 * M * LAT;
        zero_f32<<<(int)((n + 255) / 256), 256, 0, stream>>>(out, n);
        long long total = E * (2 * LAT / 4);
        spmm_scatter_split<<<(int)((total + 255) / 256), 256, 0, stream>>>(
            src, dst, ew, Z, out, total, M);
        bias_out<<<(int)((n + 255) / 256), 256, 0, stream>>>(out, bmu, blv, M * LAT);
    }
}
